// GRUActorCriticNetwork_42743514529978
// MI455X (gfx1250) — matrix operand order repaired and checked by reading, compiled, not run on hardware
//
#include <hip/hip_runtime.h>
#include <hip/hip_bf16.h>
#include <stdint.h>

// ---------------------------------------------------------------------------
// Types for CDNA5 WMMA (wave32): D(16x16 f32) = A(16x32 bf16) * B(32x16 bf16) + C
// ---------------------------------------------------------------------------
typedef __attribute__((ext_vector_type(16))) __bf16 v16bf;
typedef __attribute__((ext_vector_type(8)))  float  v8f;

union Frag {
    v16bf v;
    uint4 q[2];
};
union Pack8 {            // 8 bf16 in one 16-byte load
    uint4  q;
    __bf16 h[8];
};

static __device__ __forceinline__ unsigned short f2bf_bits(float f) {
    unsigned u = __float_as_uint(f);
    return (unsigned short)((u + 0x7FFFu + ((u >> 16) & 1u)) >> 16);  // RNE
}
static __device__ __forceinline__ __bf16 f2bf(float f) {
    unsigned short s = f2bf_bits(f);
    return __builtin_bit_cast(__bf16, s);
}
static __device__ __forceinline__ float bf2f(__bf16 x) {
    unsigned short s = __builtin_bit_cast(unsigned short, x);
    return __uint_as_float(((unsigned)s) << 16);
}
static __device__ __forceinline__ float lo_bf(uint32_t p) {   // bf16 in [15:0]
    return __uint_as_float(p << 16);
}
static __device__ __forceinline__ float hi_bf(uint32_t p) {   // bf16 in [31:16]
    return __uint_as_float(p & 0xFFFF0000u);
}

// ---------------------------------------------------------------------------
// Problem constants
// ---------------------------------------------------------------------------
#define LROLL 16384
#define HIN   512
#define NHID  1024
#define G3    3072      // 3*NHID
#define AH_W  512
#define CH_W  512
#define AACT  64

#define SCAN_BLOCKS  8
#define SCAN_THREADS 128

// ---------------------------------------------------------------------------
// f32 -> bf16 bulk convert
// ---------------------------------------------------------------------------
__global__ void k_f32_to_bf16(const float* __restrict__ src,
                              __bf16* __restrict__ dst, int n) {
    int i = blockIdx.x * blockDim.x + threadIdx.x;
    if (i < n) dst[i] = f2bf(src[i]);
}

// W_hh (3072 x 1024 f32, row-major) -> k-pair-packed transposed bf16:
// Wp[kp*3072 + j] = { bf16(W[j][2kp]) , bf16(W[j][2kp+1]) << 16 }
__global__ void k_pack_whh(const float* __restrict__ W, uint32_t* __restrict__ Wp) {
    int j  = blockIdx.x * blockDim.x + threadIdx.x;   // 0..3071 (gate-row)
    int kp = blockIdx.y;                              // 0..511  (k pair)
    if (j < G3) {
        const float* wr = W + (size_t)j * NHID + 2 * kp;
        uint32_t lo = f2bf_bits(wr[0]);
        uint32_t hi = f2bf_bits(wr[1]);
        Wp[(size_t)kp * G3 + j] = lo | (hi << 16);
    }
}

// Init: h0 = x, barrier counter = 0
__global__ void k_init(const float* __restrict__ x, float* __restrict__ hbuf,
                       int* __restrict__ bar) {
    int i = blockIdx.x * blockDim.x + threadIdx.x;
    if (i < NHID) hbuf[i] = x[i];
    if (i == 0) *bar = 0;
}

// ---------------------------------------------------------------------------
// WMMA GEMM: C[m,n] = act( sum_k A[m,k]*B[n,k] + bias[n] )
// A: M x K bf16 row-major, B: Nn x K bf16 row-major (C = A * B^T).
// Register-blocked: one wave owns a 16(M) x 64(N) strip = 4 accumulators,
// A fragment loaded once per K-chunk and reused for 4 WMMAs (10 b128 loads
// per 4 WMMAs). Block = 256 thr = 8 waves covering 512 output columns.
// Nn must be a multiple of 64 (3072 / 512 / 64 here).
// ---------------------------------------------------------------------------
__global__ void k_wmma_gemm(const __bf16* __restrict__ A,
                            const __bf16* __restrict__ B,
                            const float* __restrict__ bias,
                            void* __restrict__ Cout,
                            int M, int Nn, int K, int relu, int out_bf16) {
    const int lane   = threadIdx.x & 31;
    const int wave   = threadIdx.x >> 5;
    const int tileM  = blockIdx.y * 16;
    const int tileN0 = blockIdx.x * 512 + wave * 64;
    if (tileN0 >= Nn) return;                // wave-uniform: EXEC stays all-ones

    const int half = lane >> 4;              // 0: lanes 0-15, 1: lanes 16-31
    const int l16  = lane & 15;

    const __bf16* arow = A + (size_t)(tileM + l16) * K;
    const __bf16* br0  = B + (size_t)(tileN0 + l16) * K;       // N-tile 0
    const __bf16* br1  = br0 + (size_t)16 * K;                 // N-tile 1
    const __bf16* br2  = br0 + (size_t)32 * K;                 // N-tile 2
    const __bf16* br3  = br0 + (size_t)48 * K;                 // N-tile 3

    v8f acc0 = {}, acc1 = {}, acc2 = {}, acc3 = {};
    for (int kk = 0; kk < K; kk += 32) {
        // A 16x32 bf16: lanes 0-15 hold K 0..15, lanes 16-31 K 16..31
        Frag fa;
        fa.q[0] = *(const uint4*)(arow + kk + (half ? 16 : 0));
        fa.q[1] = *(const uint4*)(arow + kk + 8 + (half ? 16 : 0));
        // B 32x16 bf16 (col n = row n of B): lanes 0-15 K 0..15, lanes 16-31 K 16..31
        const int bo = kk + (half ? 16 : 0);
        Frag fb0, fb1, fb2, fb3;
        fb0.q[0] = *(const uint4*)(br0 + bo); fb0.q[1] = *(const uint4*)(br0 + bo + 8);
        fb1.q[0] = *(const uint4*)(br1 + bo); fb1.q[1] = *(const uint4*)(br1 + bo + 8);
        fb2.q[0] = *(const uint4*)(br2 + bo); fb2.q[1] = *(const uint4*)(br2 + bo + 8);
        fb3.q[0] = *(const uint4*)(br3 + bo); fb3.q[1] = *(const uint4*)(br3 + bo + 8);
        acc0 = __builtin_amdgcn_wmma_f32_16x16x32_bf16(false, fa.v, false, fb0.v, (short)0, acc0, false, false);
        acc1 = __builtin_amdgcn_wmma_f32_16x16x32_bf16(false, fa.v, false, fb1.v, (short)0, acc1, false, false);
        acc2 = __builtin_amdgcn_wmma_f32_16x16x32_bf16(false, fa.v, false, fb2.v, (short)0, acc2, false, false);
        acc3 = __builtin_amdgcn_wmma_f32_16x16x32_bf16(false, fa.v, false, fb3.v, (short)0, acc3, false, false);
    }

    // C/D layout: VGPR g -> (M = tileM+g+8*half, N = tileN+l16)
    v8f accs[4] = {acc0, acc1, acc2, acc3};
    #pragma unroll
    for (int tn = 0; tn < 4; ++tn) {
        int col = tileN0 + tn * 16 + l16;
        float b = bias[col];
        #pragma unroll
        for (int g = 0; g < 8; ++g) {
            int row = tileM + g + half * 8;
            float v = accs[tn][g] + b;
            if (relu) v = v > 0.f ? v : 0.f;
            if (out_bf16)
                ((__bf16*)Cout)[(size_t)row * Nn + col] = f2bf(v);
            else
                ((float*)Cout)[(size_t)row * Nn + col] = v;
        }
    }
}

// ---------------------------------------------------------------------------
// Persistent GRU scan. 8 blocks x 128 threads = 1024 threads, one hidden unit
// each. Wp is k-pair-packed (512 x 3072 dwords): per k-pair one coalesced
// dword load per gate per thread (2 bf16 / load) -> 1536 loads/step/thread.
// h broadcast via LDS (float2 reads); per-step grid barrier via global atomic.
// W_hh bf16 = 6 MB: resident in the 192 MB L2 after step 0, so each of the
// 16384 dependent steps streams from L2, not HBM.
// ---------------------------------------------------------------------------
__global__ void k_scan(const __bf16*   __restrict__ gi,     // L x 3072 (b_ih already added)
                       const uint32_t* __restrict__ Wp,     // 512 x 3072 packed k-pairs
                       const float*    __restrict__ b_hh,   // 3072
                       float*          __restrict__ hbuf,   // 1024 (init = x)
                       __bf16*         __restrict__ ybf,    // L x 1024
                       float*          __restrict__ h_final,// 1024 (into d_out)
                       int*            __restrict__ bar,
                       int L) {
    __shared__ float hs[NHID];
    const int i = blockIdx.x * SCAN_THREADS + threadIdx.x;   // 0..1023
    const float bhr = b_hh[i], bhz = b_hh[NHID + i], bhn = b_hh[2 * NHID + i];
    float hnew = 0.f;

    for (int t = 0; t < L; ++t) {
        for (int k = threadIdx.x; k < NHID; k += SCAN_THREADS) hs[k] = hbuf[k];
        __syncthreads();

        float ar = 0.f, az = 0.f, an = 0.f;
        const float2* h2 = (const float2*)hs;
        for (int kp = 0; kp < NHID / 2; ++kp) {
            const uint32_t* wr = Wp + (size_t)kp * G3;
            float2 h = h2[kp];                               // ds_load_b64
            uint32_t wr_r = wr[i];
            uint32_t wr_z = wr[NHID + i];
            uint32_t wr_n = wr[2 * NHID + i];
            ar = fmaf(lo_bf(wr_r), h.x, ar); ar = fmaf(hi_bf(wr_r), h.y, ar);
            az = fmaf(lo_bf(wr_z), h.x, az); az = fmaf(hi_bf(wr_z), h.y, az);
            an = fmaf(lo_bf(wr_n), h.x, an); an = fmaf(hi_bf(wr_n), h.y, an);
        }

        const __bf16* git = gi + (size_t)t * G3;
        float hr = ar + bhr, hz = az + bhz, hn = an + bhn;
        float r = 1.f / (1.f + __expf(-(bf2f(git[i]) + hr)));
        float z = 1.f / (1.f + __expf(-(bf2f(git[NHID + i]) + hz)));
        float n = tanhf(bf2f(git[2 * NHID + i]) + r * hn);
        float hold = hs[i];
        hnew = (1.f - z) * n + z * hold;

        hbuf[i] = hnew;
        ybf[(size_t)t * NHID + i] = f2bf(hnew);
        if (t + 1 < L)
            __builtin_prefetch(gi + (size_t)(t + 1) * G3 + i, 0, 1); // global_prefetch_b8

        // device-wide barrier (8 co-resident blocks)
        __threadfence();
        __syncthreads();
        if (threadIdx.x == 0) {
            atomicAdd(bar, 1);
            int target = SCAN_BLOCKS * (t + 1);
            while (__hip_atomic_load(bar, __ATOMIC_RELAXED, __HIP_MEMORY_SCOPE_AGENT) < target)
                __builtin_amdgcn_s_sleep(1);
        }
        __syncthreads();
        __threadfence();
    }
    h_final[i] = hnew;
}

// ---------------------------------------------------------------------------
// value[m] = c1[m,:] . Wc2 + bc2   (Nn = 1: WMMA would waste 15/16 lanes)
// uint4-vectorized: 8 bf16 per load, 64 loads per row.
// ---------------------------------------------------------------------------
__global__ void k_value(const __bf16* __restrict__ c1, const __bf16* __restrict__ w,
                        const float* __restrict__ bptr, float* __restrict__ out, int L) {
    int m = blockIdx.x * blockDim.x + threadIdx.x;
    if (m >= L) return;
    const uint4* row = (const uint4*)(c1 + (size_t)m * CH_W);
    const uint4* wv  = (const uint4*)w;
    float acc = bptr[0];
    for (int q = 0; q < CH_W / 8; ++q) {
        Pack8 a, b;
        a.q = row[q];
        b.q = wv[q];
        #pragma unroll
        for (int e = 0; e < 8; ++e)
            acc = fmaf(bf2f(a.h[e]), bf2f(b.h[e]), acc);
    }
    out[m] = acc;
}

// ---------------------------------------------------------------------------
// Host-side launcher
// ---------------------------------------------------------------------------
extern "C" void kernel_launch(void* const* d_in, const int* in_sizes, int n_in,
                              void* d_out, int out_size, void* d_ws, size_t ws_size,
                              hipStream_t stream) {
    (void)in_sizes; (void)n_in; (void)out_size; (void)ws_size;

    const float* u    = (const float*)d_in[0];
    const float* x    = (const float*)d_in[1];
    /* d_in[2] = d (episode resets) — unused by the reference */
    const float* W_ih = (const float*)d_in[3];
    const float* W_hh = (const float*)d_in[4];
    const float* b_ih = (const float*)d_in[5];
    const float* b_hh = (const float*)d_in[6];
    const float* Wa0  = (const float*)d_in[7];  const float* ba0 = (const float*)d_in[8];
    const float* Wa1  = (const float*)d_in[9];  const float* ba1 = (const float*)d_in[10];
    const float* Wa2  = (const float*)d_in[11]; const float* ba2 = (const float*)d_in[12];
    const float* Wc0  = (const float*)d_in[13]; const float* bc0 = (const float*)d_in[14];
    const float* Wc1  = (const float*)d_in[15]; const float* bc1 = (const float*)d_in[16];
    const float* Wc2  = (const float*)d_in[17]; const float* bc2 = (const float*)d_in[18];

    float* out_action = (float*)d_out;                              // L x 64
    float* out_value  = out_action + (size_t)LROLL * AACT;          // L x 1
    float* out_hfinal = out_value + LROLL;                          // 1024

    // Workspace carving (256B aligned)
    char* ws = (char*)d_ws;
    size_t off = 0;
    auto alloc = [&](size_t bytes) -> void* {
        void* p = ws + off;
        off = (off + bytes + 255) & ~((size_t)255);
        return p;
    };
    __bf16*   u_bf   = (__bf16*)alloc((size_t)LROLL * HIN * 2);
    __bf16*   Wih_bf = (__bf16*)alloc((size_t)G3 * HIN * 2);
    __bf16*   gi_bf  = (__bf16*)alloc((size_t)LROLL * G3 * 2);
    uint32_t* Whh_p  = (uint32_t*)alloc((size_t)(NHID / 2) * G3 * 4);
    __bf16*   y_bf   = (__bf16*)alloc((size_t)LROLL * NHID * 2);
    __bf16*   a0_bf  = (__bf16*)alloc((size_t)LROLL * AH_W * 2);
    __bf16*   a1_bf  = (__bf16*)alloc((size_t)LROLL * AH_W * 2);
    __bf16*   c0_bf  = (__bf16*)alloc((size_t)LROLL * CH_W * 2);
    __bf16*   c1_bf  = (__bf16*)alloc((size_t)LROLL * CH_W * 2);
    __bf16*   Wa0_bf = (__bf16*)alloc((size_t)AH_W * NHID * 2);
    __bf16*   Wa1_bf = (__bf16*)alloc((size_t)AH_W * AH_W * 2);
    __bf16*   Wa2_bf = (__bf16*)alloc((size_t)AACT * AH_W * 2);
    __bf16*   Wc0_bf = (__bf16*)alloc((size_t)CH_W * NHID * 2);
    __bf16*   Wc1_bf = (__bf16*)alloc((size_t)CH_W * CH_W * 2);
    __bf16*   Wc2_bf = (__bf16*)alloc((size_t)CH_W * 2);
    float*    hbuf   = (float*)alloc(NHID * 4);
    int*      bar    = (int*)alloc(256);

    auto conv = [&](const float* src, __bf16* dst, size_t n) {
        k_f32_to_bf16<<<dim3((unsigned)((n + 255) / 256)), dim3(256), 0, stream>>>(src, dst, (int)n);
    };

    // --- Stage 0: precision staging -------------------------------------
    conv(u,    u_bf,   (size_t)LROLL * HIN);
    conv(W_ih, Wih_bf, (size_t)G3 * HIN);
    conv(Wa0,  Wa0_bf, (size_t)AH_W * NHID);
    conv(Wa1,  Wa1_bf, (size_t)AH_W * AH_W);
    conv(Wa2,  Wa2_bf, (size_t)AACT * AH_W);
    conv(Wc0,  Wc0_bf, (size_t)CH_W * NHID);
    conv(Wc1,  Wc1_bf, (size_t)CH_W * CH_W);
    conv(Wc2,  Wc2_bf, (size_t)CH_W);
    // W_hh -> k-pair-packed transposed bf16 for coalesced dword matvec loads
    k_pack_whh<<<dim3(G3 / 256, NHID / 2), dim3(256), 0, stream>>>(W_hh, Whh_p);
    k_init<<<dim3(4), dim3(256), 0, stream>>>(x, hbuf, bar);

    // --- Stage 1: gi = u @ W_ih^T + b_ih  (WMMA bf16) --------------------
    k_wmma_gemm<<<dim3(G3 / 512, LROLL / 16), dim3(256), 0, stream>>>(
        u_bf, Wih_bf, b_ih, gi_bf, LROLL, G3, HIN, /*relu=*/0, /*bf16=*/1);

    // --- Stage 2: sequential GRU scan (persistent, grid barrier) ---------
    k_scan<<<dim3(SCAN_BLOCKS), dim3(SCAN_THREADS), 0, stream>>>(
        gi_bf, Whh_p, b_hh, hbuf, y_bf, out_hfinal, bar, LROLL);

    // --- Stage 3: actor head (WMMA bf16) ---------------------------------
    k_wmma_gemm<<<dim3(1, LROLL / 16), dim3(256), 0, stream>>>(
        y_bf, Wa0_bf, ba0, a0_bf, LROLL, AH_W, NHID, 1, 1);
    k_wmma_gemm<<<dim3(1, LROLL / 16), dim3(256), 0, stream>>>(
        a0_bf, Wa1_bf, ba1, a1_bf, LROLL, AH_W, AH_W, 1, 1);
    k_wmma_gemm<<<dim3(1, LROLL / 16), dim3(256), 0, stream>>>(
        a1_bf, Wa2_bf, ba2, out_action, LROLL, AACT, AH_W, 0, 0);

    // --- Stage 4: critic head --------------------------------------------
    k_wmma_gemm<<<dim3(1, LROLL / 16), dim3(256), 0, stream>>>(
        y_bf, Wc0_bf, bc0, c0_bf, LROLL, CH_W, NHID, 1, 1);
    k_wmma_gemm<<<dim3(1, LROLL / 16), dim3(256), 0, stream>>>(
        c0_bf, Wc1_bf, bc1, c1_bf, LROLL, CH_W, CH_W, 1, 1);
    k_value<<<dim3(LROLL / 256), dim3(256), 0, stream>>>(
        c1_bf, Wc2_bf, bc2, out_value, LROLL);
}
